// MultiHeadAttention_38654705664243
// MI455X (gfx1250) — compile-verified, matrix operand order checked by reading, not run on hardware
//
#include <hip/hip_runtime.h>
#include <hip/hip_bf16.h>

// ---------------------------------------------------------------------------
// MHA (B=16, T=1024, E=1024, H=16, Dh=64) for gfx1250 (MI455X), wave32.
// bf16 WMMA (v_wmma_f32_16x16x32_bf16) with fp32 accumulation everywhere.
// Flash-attention (online softmax) so the 1GB (fp32) B*H*T*T score matrix is never formed.
// Pure-copy LDS staging uses the CDNA5 async DMA path (ASYNCcnt).
// ---------------------------------------------------------------------------

static constexpr int Bz = 16, Tn = 1024, En = 1024, Hn = 16, Dhn = 64;

typedef __attribute__((ext_vector_type(16))) __bf16 bf16x16;
typedef __attribute__((ext_vector_type(8)))  __bf16 bf16x8;
typedef __attribute__((ext_vector_type(8)))  float  f32x8;

// Native f32->bf16 (RNE) — lowers to v_cvt_pk_bf16_f32 on gfx1250.
__device__ __forceinline__ __bf16 bf16cast(float f) { return (__bf16)f; }

// CDNA5 async global->LDS 16-byte copy (GLOBAL_LOAD_ASYNC_TO_LDS_B128, GV mode).
// LDS dest address = low 32 bits of the flat pointer.
__device__ __forceinline__ void async_copy_b128(__bf16* lds_dst, const __bf16* gsrc) {
  unsigned lds = (unsigned)(unsigned long long)lds_dst;
  asm volatile("global_load_async_to_lds_b128 %0, %1, off"
               :: "v"(lds), "v"(gsrc) : "memory");
}
__device__ __forceinline__ void wait_async0() {
  asm volatile("s_wait_asynccnt 0x0" ::: "memory");
}

// Load one A/B fragment row for 16x16x32 bf16 WMMA.
// Per ISA 7.12.2 (16-bit A 16x32): lane (idx%16 = m/n, half = lane/16) holds
// K chunks [half*8, half*8+8) and [16+half*8, 16+half*8+8): two aligned b128s.
__device__ __forceinline__ bf16x16 load_frag_row32(const __bf16* row, int half) {
  bf16x8 lo = *(const bf16x8*)(row + half * 8);
  bf16x8 hi = *(const bf16x8*)(row + 16 + half * 8);
  bf16x16 r;
#pragma unroll
  for (int i = 0; i < 8; ++i) { r[i] = lo[i]; r[i + 8] = hi[i]; }
  return r;
}

__device__ __forceinline__ f32x8 wmma_bf16(bf16x16 a, bf16x16 b, f32x8 c) {
  return __builtin_amdgcn_wmma_f32_16x16x32_bf16(
      /*neg_a=*/false, a, /*neg_b=*/false, b,
      /*c_mod=*/(short)0, c, /*reuse_a=*/false, /*reuse_b=*/false);
}

// ---------------------------------------------------------------------------
// Kernel 1: QKV projection.  [B*T,E] x [H,E,Dh] -> bf16 [B,H,T,Dh] x3.
// 8 waves, 128-row x 64-col tile per WG; x tile reused for Wq/Wk/Wv.
// ---------------------------------------------------------------------------
__global__ __launch_bounds__(256) void qkv_proj_kernel(
    const float* __restrict__ x,
    const float* __restrict__ Wq, const float* __restrict__ Wk,
    const float* __restrict__ Wv,
    __bf16* __restrict__ Qb, __bf16* __restrict__ Kb, __bf16* __restrict__ Vb) {
  const int h    = blockIdx.y;
  const int row0 = blockIdx.x * 128;           // flat (b*T + t)
  const int tid  = threadIdx.x;
  const int wave = tid >> 5, lane = tid & 31;
  const int lm   = lane & 15, half = lane >> 4;

  __shared__ __align__(16) __bf16 lA[128][32];       // x tile (bf16)
  __shared__ __align__(16) __bf16 lB[3][64][32];     // W^T tiles: [n=d][k=e]

  f32x8 acc[3][4] = {};

  for (int k0 = 0; k0 < En; k0 += 32) {
    { // stage x: one contiguous half-row (16 f32) per thread, packed converts
      const int r = tid >> 1, c0 = (tid & 1) * 16;
      const float* gx = x + (size_t)(row0 + r) * En + k0 + c0;
      bf16x8 p0, p1;
#pragma unroll
      for (int j = 0; j < 8; ++j) {
        p0[j] = bf16cast(gx[j]);
        p1[j] = bf16cast(gx[8 + j]);
      }
      *(bf16x8*)&lA[r][c0]     = p0;
      *(bf16x8*)&lA[r][c0 + 8] = p1;
    }
#pragma unroll
    for (int i = 0; i < 8; ++i) {                    // stage 3 weight tiles
      const int idx = tid + i * 256;                 // 0..2047
      const int d = idx & 63, c = idx >> 6;          // coalesced over d
      const size_t g = ((size_t)h * En + (size_t)(k0 + c)) * Dhn + d;
      lB[0][d][c] = bf16cast(Wq[g]);
      lB[1][d][c] = bf16cast(Wk[g]);
      lB[2][d][c] = bf16cast(Wv[g]);
    }
    __syncthreads();
    const bf16x16 a = load_frag_row32(&lA[wave * 16 + lm][0], half);
#pragma unroll
    for (int ct = 0; ct < 4; ++ct) {
      const bf16x16 bq = load_frag_row32(&lB[0][ct * 16 + lm][0], half);
      const bf16x16 bk = load_frag_row32(&lB[1][ct * 16 + lm][0], half);
      const bf16x16 bv = load_frag_row32(&lB[2][ct * 16 + lm][0], half);
      acc[0][ct] = wmma_bf16(a, bq, acc[0][ct]);
      acc[1][ct] = wmma_bf16(a, bk, acc[1][ct]);
      acc[2][ct] = wmma_bf16(a, bv, acc[2][ct]);
    }
    __syncthreads();
  }

  const int b  = row0 / Tn;
  const int t0 = (row0 % Tn) + wave * 16;
  const size_t base = ((size_t)b * Hn + h) * Tn;
  __bf16* outs[3] = {Qb, Kb, Vb};
#pragma unroll
  for (int m = 0; m < 3; ++m)
#pragma unroll
    for (int ct = 0; ct < 4; ++ct)
#pragma unroll
      for (int r = 0; r < 8; ++r) {
        const int t = t0 + r + 8 * half;             // C/D layout: m = r + 8*half
        const int d = ct * 16 + lm;                  // n = lane%16
        outs[m][(base + t) * Dhn + d] = bf16cast(acc[m][ct][r]);
      }
}

// ---------------------------------------------------------------------------
// Kernel 2: causal flash attention.  4 waves/WG; wave = one 16-row Q block;
// WG shares K/V staging for 4 adjacent Q blocks of one (b,h).
// ---------------------------------------------------------------------------
__global__ __launch_bounds__(128) void attn_kernel(
    const __bf16* __restrict__ Qb, const __bf16* __restrict__ Kb,
    const __bf16* __restrict__ Vb, __bf16* __restrict__ Oc) {
  const int bh = blockIdx.y;
  const int b = bh >> 4, h = bh & 15;                // H = 16
  const int tid = threadIdx.x, wave = tid >> 5, lane = tid & 31;
  const int lm = lane & 15, half = lane >> 4;
  const int q0   = blockIdx.x * 64 + wave * 16;
  const int smax = blockIdx.x * 64 + 64;             // exclusive key bound for WG

  const __bf16* Qp = Qb + (size_t)bh * Tn * Dhn;
  const __bf16* Kp = Kb + (size_t)bh * Tn * Dhn;
  const __bf16* Vp = Vb + (size_t)bh * Tn * Dhn;

  __shared__ __align__(16) __bf16 lK[32][64];        // K rows (B-frag rows for QK^T)
  __shared__ __align__(16) __bf16 lVt[64][32];       // V^T rows (B-frag rows for P*V)
  __shared__ __align__(16) __bf16 lP[4][16][32];     // per-wave P relayout patch

  // Q fragments once per wave: A = Q block 16x64 -> two K-windows of 32
  const bf16x16 aq0 = load_frag_row32(Qp + (size_t)(q0 + lm) * Dhn, half);
  const bf16x16 aq1 = load_frag_row32(Qp + (size_t)(q0 + lm) * Dhn + 32, half);

  f32x8 o[4] = {};
  float rmax[8], rsum[8];
#pragma unroll
  for (int r = 0; r < 8; ++r) { rmax[r] = -__builtin_inff(); rsum[r] = 0.f; }

  constexpr float scalef = 0.125f;                   // 1/sqrt(64)

  for (int s0 = 0; s0 < smax; s0 += 32) {
    { // stage K tile [32][64] via CDNA5 async DMA (no VGPR round-trip)
      const int r = tid >> 2, c = (tid & 3) * 16;
      const __bf16* g = Kp + (size_t)(s0 + r) * Dhn + c;
      async_copy_b128(&lK[r][c],     g);
      async_copy_b128(&lK[r][c + 8], g + 8);
    }
#pragma unroll
    for (int i = 0; i < 16; ++i) {                   // stage V transposed
      const int idx = tid + i * 128;
      const int r = idx >> 6, c = idx & 63;          // coalesced global read
      lVt[c][r] = Vp[(size_t)(s0 + r) * Dhn + c];
    }
    if (s0 + 32 < smax)                               // CDNA5 global_prefetch_b8
      __builtin_prefetch(Kp + (size_t)(s0 + 32) * Dhn + (tid & 63), 0, 0);
    wait_async0();                                    // K tile resident in LDS
    __syncthreads();

    if (s0 < q0 + 16) {                              // wave-uniform (EXEC stays full)
      // S = scale * Q K^T for 32 keys -> two 16x16 tiles
      f32x8 sl = {}, sh = {};
      {
        const bf16x16 b0 = load_frag_row32(&lK[lm][0], half);
        sl = wmma_bf16(aq0, b0, sl);
        const bf16x16 b1 = load_frag_row32(&lK[lm][32], half);
        sl = wmma_bf16(aq1, b1, sl);
        const bf16x16 b2 = load_frag_row32(&lK[16 + lm][0], half);
        sh = wmma_bf16(aq0, b2, sh);
        const bf16x16 b3 = load_frag_row32(&lK[16 + lm][32], half);
        sh = wmma_bf16(aq1, b3, sh);
      }
      // online softmax; row m = r + 8*half lives in one 16-lane half
#pragma unroll
      for (int r = 0; r < 8; ++r) {
        const int t = q0 + r + 8 * half;
        float vl = sl[r] * scalef, vh = sh[r] * scalef;
        if (s0 + lm > t)      vl = -__builtin_inff();
        if (s0 + 16 + lm > t) vh = -__builtin_inff();
        float m = fmaxf(vl, vh);
#pragma unroll
        for (int off = 1; off < 16; off <<= 1) m = fmaxf(m, __shfl_xor(m, off, 32));
        const float nm    = fmaxf(rmax[r], m);
        const float alpha = __expf(rmax[r] - nm);    // first iter: exp(-inf)=0
        const float pl = __expf(vl - nm), ph = __expf(vh - nm);
        float ls = pl + ph;
#pragma unroll
        for (int off = 1; off < 16; off <<= 1) ls += __shfl_xor(ls, off, 32);
        rsum[r] = rsum[r] * alpha + ls;
        rmax[r] = nm;
#pragma unroll
        for (int ct = 0; ct < 4; ++ct) o[ct][r] *= alpha;
        const int mrow = r + 8 * half;               // D-layout -> LDS (A-layout patch)
        lP[wave][mrow][lm]      = bf16cast(pl);
        lP[wave][mrow][16 + lm] = bf16cast(ph);
      }
      // O += P * V  (LDS is in-order per wave: stores above precede these loads)
      const bf16x16 ap = load_frag_row32(&lP[wave][lm][0], half);
#pragma unroll
      for (int ct = 0; ct < 4; ++ct) {
        const bf16x16 bv = load_frag_row32(&lVt[ct * 16 + lm][0], half);
        o[ct] = wmma_bf16(ap, bv, o[ct]);
      }
    }
    __syncthreads();
  }

  // finalize and store concat-head layout [B, T, H*Dh] (bf16)
#pragma unroll
  for (int r = 0; r < 8; ++r) {
    const int t = q0 + r + 8 * half;
    const float inv = 1.f / rsum[r];
#pragma unroll
    for (int ct = 0; ct < 4; ++ct) {
      const int d = ct * 16 + lm;
      Oc[((size_t)b * Tn + t) * (size_t)(Hn * Dhn) + h * Dhn + d] =
          bf16cast(o[ct][r] * inv);
    }
  }
}

// ---------------------------------------------------------------------------
// Kernel 3: output projection.  [B*T,1024](bf16) x Wo[1024,1024] + bo -> f32.
// ---------------------------------------------------------------------------
__global__ __launch_bounds__(256) void out_proj_kernel(
    const __bf16* __restrict__ Oc, const float* __restrict__ Wo,
    const float* __restrict__ bo, float* __restrict__ out) {
  const int n0   = blockIdx.x * 64;
  const int row0 = blockIdx.y * 128;
  const int tid = threadIdx.x, wave = tid >> 5, lane = tid & 31;
  const int lm = lane & 15, half = lane >> 4;

  __shared__ __align__(16) __bf16 lA[128][32];
  __shared__ __align__(16) __bf16 lB[64][32];        // Wo^T tile: [n][k]

  f32x8 acc[4] = {};
  constexpr int K = Hn * Dhn;                        // 1024
  for (int k0 = 0; k0 < K; k0 += 32) {
    { // A already bf16: async DMA global->LDS copies
      const int r = tid >> 1, c = (tid & 1) * 16;
      const __bf16* g = Oc + (size_t)(row0 + r) * K + k0 + c;
      async_copy_b128(&lA[r][c],     g);
      async_copy_b128(&lA[r][c + 8], g + 8);
    }
#pragma unroll
    for (int i = 0; i < 8; ++i) {                    // stage+convert+transpose Wo
      const int idx = tid + i * 256;                 // 0..2047
      const int k = idx >> 6, n = idx & 63;          // coalesced over n
      lB[n][k] = bf16cast(Wo[(size_t)(k0 + k) * En + n0 + n]);
    }
    wait_async0();                                   // A tile resident in LDS
    __syncthreads();
    const bf16x16 a = load_frag_row32(&lA[wave * 16 + lm][0], half);
#pragma unroll
    for (int ct = 0; ct < 4; ++ct) {
      const bf16x16 bb = load_frag_row32(&lB[ct * 16 + lm][0], half);
      acc[ct] = wmma_bf16(a, bb, acc[ct]);
    }
    __syncthreads();
  }
#pragma unroll
  for (int ct = 0; ct < 4; ++ct)
#pragma unroll
    for (int r = 0; r < 8; ++r) {
      const int row = row0 + wave * 16 + r + 8 * half;
      const int n = n0 + ct * 16 + lm;
      out[(size_t)row * En + n] = acc[ct][r] + bo[n];
    }
}

// ---------------------------------------------------------------------------
extern "C" void kernel_launch(void* const* d_in, const int* in_sizes, int n_in,
                              void* d_out, int out_size, void* d_ws, size_t ws_size,
                              hipStream_t stream) {
  const float* x  = (const float*)d_in[0];
  const float* Wq = (const float*)d_in[1];
  const float* Wk = (const float*)d_in[2];
  const float* Wv = (const float*)d_in[3];
  const float* Wo = (const float*)d_in[4];
  const float* bo = (const float*)d_in[5];
  float* out = (float*)d_out;

  // workspace: Q,K,V bf16 [B,H,T,Dh] + O bf16 [B,T,H*Dh]  (4 x 32 MB)
  const size_t nQ = (size_t)Bz * Hn * Tn * Dhn;
  __bf16* Qb = (__bf16*)d_ws;
  __bf16* Kb = Qb + nQ;
  __bf16* Vb = Kb + nQ;
  __bf16* Oc = Vb + nQ;

  qkv_proj_kernel<<<dim3((Bz * Tn) / 128, Hn), 256, 0, stream>>>(
      x, Wq, Wk, Wv, Qb, Kb, Vb);
  attn_kernel<<<dim3(Tn / 64, Bz * Hn), 128, 0, stream>>>(Qb, Kb, Vb, Oc);
  out_proj_kernel<<<dim3(En / 64, (Bz * Tn) / 128), 256, 0, stream>>>(
      Oc, Wo, bo, out);
}
